// LayerNormLSTM_18691697672920
// MI455X (gfx1250) — matrix operand order repaired and checked by reading, not compiled, not run on hardware
//
#include <hip/hip_runtime.h>
#include <hip/hip_bf16.h>
#include <stdint.h>

// ---------------------------------------------------------------------------
// LayerNorm-LSTM (2 layers) for MI455X / gfx1250, wave32 + WMMA bf16.
//   B=512, T=512, I=128, H=512, gates G=2048, classes C=6.
// Strategy: batch rows are independent through the recurrence -> 32 persistent
// workgroups x 16 batch rows. h kept in LDS (bf16), c kept in registers
// (wave w owns batch row w). Weights pre-transposed+converted to bf16 in
// workspace as Wt[N][Kconcat] so WMMA B-fragments are contiguous 32B loads.
// ---------------------------------------------------------------------------

typedef __bf16 bf16_t;
typedef bf16_t v8bf  __attribute__((ext_vector_type(8)));
typedef bf16_t v16bf __attribute__((ext_vector_type(16)));
typedef float  v8f   __attribute__((ext_vector_type(8)));

#define BB   512
#define TT   512
#define II   128
#define HH   512
#define GG   2048   // 4*H
#define NC   6
#define K0   640    // I + H   (layer0 concat K)
#define K1   1024   // H + H   (layer1 concat K)
#define XS   136    // x LDS row stride (bf16 elems), padded
#define HS   520    // h LDS row stride (bf16 elems), padded
#define ROWS 16
#define LNEPS 1e-5f

__device__ __forceinline__ bf16_t f2bf(float f) {
  union { float f; uint32_t u; } v; v.f = f;
  uint32_t r = v.u + 0x7FFFu + ((v.u >> 16) & 1u);   // round-to-nearest-even
  uint16_t h = (uint16_t)(r >> 16);
  bf16_t o; __builtin_memcpy(&o, &h, 2); return o;
}
__device__ __forceinline__ float bf2f(bf16_t b) {
  uint16_t h; __builtin_memcpy(&h, &b, 2);
  union { uint32_t u; float f; } v; v.u = ((uint32_t)h) << 16; return v.f;
}
__device__ __forceinline__ float sigm(float x) { return 1.f / (1.f + __expf(-x)); }

// ---------------------------------------------------------------------------
// One-time weight prep: bf16 convert + transpose + concat along K.
//   W0t[n][k] : k in [0,128) = Wx0[k][n], k in [128,640)  = Wh0[k-128][n]
//   W1t[n][k] : k in [0,512) = Wx1[k][n], k in [512,1024) = Wh1[k-512][n]
// ---------------------------------------------------------------------------
__global__ __launch_bounds__(256)
void convert_weights(const float* __restrict__ Wx0, const float* __restrict__ Wh0,
                     const float* __restrict__ Wx1, const float* __restrict__ Wh1,
                     bf16_t* __restrict__ W0t, bf16_t* __restrict__ W1t) {
  const int n = blockIdx.x;                       // gate column 0..2047
  for (int k = threadIdx.x; k < K0; k += 256) {
    float v = (k < II) ? Wx0[(size_t)k * GG + n] : Wh0[(size_t)(k - II) * GG + n];
    W0t[(size_t)n * K0 + k] = f2bf(v);
  }
  for (int k = threadIdx.x; k < K1; k += 256) {
    float v = (k < HH) ? Wx1[(size_t)k * GG + n] : Wh1[(size_t)(k - HH) * GG + n];
    W1t[(size_t)n * K1 + k] = f2bf(v);
  }
}

// ---------------------------------------------------------------------------
// Persistent fused LSTM: one block = 16 batch rows through all 512 steps.
// 512 threads = 16 waves. GEMM: wave covers 8 N-tiles of 16 gate columns.
// ---------------------------------------------------------------------------
__global__ __launch_bounds__(512)
void lstm_fused(const float* __restrict__ x,
                const bf16_t* __restrict__ W0t, const bf16_t* __restrict__ W1t,
                const float* __restrict__ bx0,  const float* __restrict__ bx1,
                const float* __restrict__ lng0, const float* __restrict__ lnb0,
                const float* __restrict__ lncg0,const float* __restrict__ lncb0,
                const float* __restrict__ lng1, const float* __restrict__ lnb1,
                const float* __restrict__ lncg1,const float* __restrict__ lncb1,
                const float* __restrict__ fcW,  const float* __restrict__ fcb,
                float* __restrict__ out) {
  __shared__ bf16_t xs[ROWS * XS];     //  4.3 KB  x_t tile (bf16)
  __shared__ bf16_t h0s[ROWS * HS];    // 16.6 KB  layer0 hidden (bf16)
  __shared__ bf16_t h1s[ROWS * HS];    // 16.6 KB  layer1 hidden (bf16)
  __shared__ float  gs[ROWS * GG];     // 128 KB   raw gates (f32)

  const int tid  = threadIdx.x;
  const int wave = tid >> 5;
  const int lane = tid & 31;
  const int b0   = blockIdx.x * ROWS;

  for (int i = tid; i < ROWS * HS; i += 512) { h0s[i] = f2bf(0.f); h1s[i] = f2bf(0.f); }

  float c0v[16], c1v[16];              // cell states live in registers
#pragma unroll
  for (int j = 0; j < 16; ++j) { c0v[j] = 0.f; c1v[j] = 0.f; }

  // WMMA lane->element mapping (16-bit A 16x32, B 32x16, C/D f32 16x16)
  const int mA    = lane & 15;               // A: row (batch)
  const int koffA = (lane < 16) ? 0 : 16;    // A: K sub-offset
  const int koffB = (lane < 16) ? 0 : 16;    // B: K sub-offset
  const int nB    = lane & 15;               // B/C: column
  const int mb    = (lane < 16) ? 0 : 8;     // C: row base per vgpr

  // Per-row LayerNorm + cell update (wave `wave` owns batch row `wave`).
  auto cell_update = [&](const float* __restrict__ lng, const float* __restrict__ lnb,
                         const float* __restrict__ lncg, const float* __restrict__ lncb,
                         float (&cst)[16], bf16_t* __restrict__ hdst) {
    const int m = wave;
    float s = 0.f, q = 0.f;
    for (int c = lane; c < GG; c += 32) { float v = gs[m * GG + c]; s += v; q += v * v; }
#pragma unroll
    for (int o = 16; o > 0; o >>= 1) { s += __shfl_xor(s, o); q += __shfl_xor(q, o); }
    const float mu = s * (1.f / GG);
    const float rs = rsqrtf(q * (1.f / GG) - mu * mu + LNEPS);

    float s2 = 0.f, q2 = 0.f, cv[16], ov[16];
#pragma unroll
    for (int jj = 0; jj < 16; ++jj) {
      const int j = lane + jj * 32;
      const float ni = (gs[m * GG + j]          - mu) * rs * lng[j]          + lnb[j];
      const float nf = (gs[m * GG + HH + j]     - mu) * rs * lng[HH + j]     + lnb[HH + j];
      const float ng = (gs[m * GG + 2 * HH + j] - mu) * rs * lng[2 * HH + j] + lnb[2 * HH + j];
      const float no = (gs[m * GG + 3 * HH + j] - mu) * rs * lng[3 * HH + j] + lnb[3 * HH + j];
      const float cn = sigm(nf) * cst[jj] + sigm(ni) * tanhf(ng);
      cv[jj] = cn; ov[jj] = no; s2 += cn; q2 += cn * cn;
    }
#pragma unroll
    for (int o = 16; o > 0; o >>= 1) { s2 += __shfl_xor(s2, o); q2 += __shfl_xor(q2, o); }
    const float mu2 = s2 * (1.f / HH);
    const float rs2 = rsqrtf(q2 * (1.f / HH) - mu2 * mu2 + LNEPS);
#pragma unroll
    for (int jj = 0; jj < 16; ++jj) {
      const int j = lane + jj * 32;
      const float cn = (cv[jj] - mu2) * rs2 * lncg[j] + lncb[j];
      cst[jj] = cn;
      hdst[m * HS + j] = f2bf(sigm(ov[jj]) * tanhf(cn));
    }
  };

  __syncthreads();

#pragma unroll 1
  for (int t = 0; t < TT; ++t) {
    // ---- stage x_t tile into LDS (f32 -> bf16), 4 elems/thread -------------
    {
      const int e = tid * 4, r = e >> 7, c = e & 127;
      const float4 xv = *(const float4*)&x[((size_t)(b0 + r) * TT + t) * II + c];
      bf16_t* dst = &xs[r * XS + c];
      dst[0] = f2bf(xv.x); dst[1] = f2bf(xv.y); dst[2] = f2bf(xv.z); dst[3] = f2bf(xv.w);
    }
    __syncthreads();

    // ---- layer 0 gate GEMM: [16 x 640] @ W0t^T -> gates[16 x 2048] ---------
#pragma unroll 1
    for (int nt = 0; nt < 8; ++nt) {
      const int ncol = (wave * 8 + nt) * 16;
      const bf16_t* wrow = W0t + (size_t)(ncol + nB) * K0 + koffB;
      __builtin_prefetch((const void*)(wrow + (size_t)16 * K0), 0, 0);  // next N-tile
      v8f acc = {0.f, 0.f, 0.f, 0.f, 0.f, 0.f, 0.f, 0.f};
#pragma unroll
      for (int kt = 0; kt < 4; ++kt) {                  // x segment: K = 0..127
        v8bf lo = *(const v8bf*)&xs[mA * XS + kt * 32 + koffA];
        v8bf hi = *(const v8bf*)&xs[mA * XS + kt * 32 + koffA + 8];
        v16bf a = __builtin_shufflevector(lo, hi, 0,1,2,3,4,5,6,7,8,9,10,11,12,13,14,15);
        v16bf b = *(const v16bf*)(wrow + kt * 32);
        acc = __builtin_amdgcn_wmma_f32_16x16x32_bf16(false, a, false, b, (short)0, acc, false, false);
      }
#pragma unroll
      for (int kt = 0; kt < 16; ++kt) {                 // h0 segment: K = 128..639
        v8bf lo = *(const v8bf*)&h0s[mA * HS + kt * 32 + koffA];
        v8bf hi = *(const v8bf*)&h0s[mA * HS + kt * 32 + koffA + 8];
        v16bf a = __builtin_shufflevector(lo, hi, 0,1,2,3,4,5,6,7,8,9,10,11,12,13,14,15);
        v16bf b = *(const v16bf*)(wrow + II + kt * 32);
        acc = __builtin_amdgcn_wmma_f32_16x16x32_bf16(false, a, false, b, (short)0, acc, false, false);
      }
      const float bxv = bx0[ncol + nB];
#pragma unroll
      for (int r = 0; r < 8; ++r) gs[(mb + r) * GG + ncol + nB] = acc[r] + bxv;
    }
    __syncthreads();

    cell_update(lng0, lnb0, lncg0, lncb0, c0v, h0s);    // writes new h0 (bf16)
    __syncthreads();

    // ---- layer 1 gate GEMM: [16 x 1024] @ W1t^T, A = [h0_new | h1_prev] ----
#pragma unroll 1
    for (int nt = 0; nt < 8; ++nt) {
      const int ncol = (wave * 8 + nt) * 16;
      const bf16_t* wrow = W1t + (size_t)(ncol + nB) * K1 + koffB;
      __builtin_prefetch((const void*)(wrow + (size_t)16 * K1), 0, 0);
      v8f acc = {0.f, 0.f, 0.f, 0.f, 0.f, 0.f, 0.f, 0.f};
#pragma unroll
      for (int kt = 0; kt < 16; ++kt) {                 // h0 segment: K = 0..511
        v8bf lo = *(const v8bf*)&h0s[mA * HS + kt * 32 + koffA];
        v8bf hi = *(const v8bf*)&h0s[mA * HS + kt * 32 + koffA + 8];
        v16bf a = __builtin_shufflevector(lo, hi, 0,1,2,3,4,5,6,7,8,9,10,11,12,13,14,15);
        v16bf b = *(const v16bf*)(wrow + kt * 32);
        acc = __builtin_amdgcn_wmma_f32_16x16x32_bf16(false, a, false, b, (short)0, acc, false, false);
      }
#pragma unroll
      for (int kt = 0; kt < 16; ++kt) {                 // h1 segment: K = 512..1023
        v8bf lo = *(const v8bf*)&h1s[mA * HS + kt * 32 + koffA];
        v8bf hi = *(const v8bf*)&h1s[mA * HS + kt * 32 + koffA + 8];
        v16bf a = __builtin_shufflevector(lo, hi, 0,1,2,3,4,5,6,7,8,9,10,11,12,13,14,15);
        v16bf b = *(const v16bf*)(wrow + HH + kt * 32);
        acc = __builtin_amdgcn_wmma_f32_16x16x32_bf16(false, a, false, b, (short)0, acc, false, false);
      }
      const float bxv = bx1[ncol + nB];
#pragma unroll
      for (int r = 0; r < 8; ++r) gs[(mb + r) * GG + ncol + nB] = acc[r] + bxv;
    }
    __syncthreads();

    cell_update(lng1, lnb1, lncg1, lncb1, c1v, h1s);    // writes new h1 (bf16)
    __syncthreads();
  }

  // ---- final FC: out[b] = h1 @ fcW + fcb  (96 outputs per block) -----------
  if (tid < ROWS * NC) {
    const int r = tid / NC, cc = tid % NC;
    float acc = fcb[cc];
    for (int j = 0; j < HH; ++j) acc += bf2f(h1s[r * HS + j]) * fcW[(size_t)j * NC + cc];
    out[(size_t)(b0 + r) * NC + cc] = acc;
  }
}

// ---------------------------------------------------------------------------
extern "C" void kernel_launch(void* const* d_in, const int* in_sizes, int n_in,
                              void* d_out, int out_size, void* d_ws, size_t ws_size,
                              hipStream_t stream) {
  const float* x     = (const float*)d_in[0];
  const float* Wx0   = (const float*)d_in[1];
  const float* bx0   = (const float*)d_in[2];
  const float* Wh0   = (const float*)d_in[3];
  const float* lng0  = (const float*)d_in[4];
  const float* lnb0  = (const float*)d_in[5];
  const float* lncg0 = (const float*)d_in[6];
  const float* lncb0 = (const float*)d_in[7];
  const float* Wx1   = (const float*)d_in[8];
  const float* bx1   = (const float*)d_in[9];
  const float* Wh1   = (const float*)d_in[10];
  const float* lng1  = (const float*)d_in[11];
  const float* lnb1  = (const float*)d_in[12];
  const float* lncg1 = (const float*)d_in[13];
  const float* lncb1 = (const float*)d_in[14];
  const float* fcW   = (const float*)d_in[15];
  const float* fcb   = (const float*)d_in[16];
  float* out = (float*)d_out;

  bf16_t* W0t = (bf16_t*)d_ws;                    // 2048 x 640  bf16 (2.6 MB)
  bf16_t* W1t = W0t + (size_t)GG * K0;            // 2048 x 1024 bf16 (4.2 MB)

  hipLaunchKernelGGL(convert_weights, dim3(GG), dim3(256), 0, stream,
                     Wx0, Wh0, Wx1, Wh1, W0t, W1t);
  hipLaunchKernelGGL(lstm_fused, dim3(BB / ROWS), dim3(512), 0, stream,
                     x, W0t, W1t, bx0, bx1,
                     lng0, lnb0, lncg0, lncb0,
                     lng1, lnb1, lncg1, lncb1,
                     fcW, fcb, out);
}
